// VersorAblation_12713103196434
// MI455X (gfx1250) — matrix operand order repaired and checked by reading, not compiled, not run on hardware
//
#include <hip/hip_runtime.h>

typedef __attribute__((ext_vector_type(16))) _Float16 v16h;
typedef __attribute__((ext_vector_type(8)))  float    v8f;

namespace {

constexpr int kS  = 256;             // sequence length
constexpr int kD  = 6;               // feature dim
constexpr int kND  = 64 * kD;        // 384  (N*D)
constexpr int kSND = kS * kND;       // 98304 (S*N*D)

// Compile-time Cayley sign table for Cl(4,1), metric (+,+,+,+,-):
// C[a][b][a^b] = sign(a,b); folded into FMA neg modifiers after unrolling.
struct SignTab {
  float s[32][32];
  constexpr SignTab() : s{} {
    for (int a = 0; a < 32; ++a)
      for (int b = 0; b < 32; ++b) {
        int par = 0;
        for (int t = a >> 1; t; t >>= 1) {
          int x = t & b;
          while (x) { par ^= (x & 1); x >>= 1; }
        }
        par ^= ((a & b) >> 4) & 1;   // e4*e4 = -1
        s[a][b] = par ? -1.0f : 1.0f;
      }
  }
};
constexpr SignTab kSign{};

} // namespace

// out = x + b_out  (atomic accumulation target for the scan kernel)
__global__ void versor_init_out(const float* __restrict__ x,
                                const float* __restrict__ b_out,
                                float* __restrict__ out, int n) {
  int i = blockIdx.x * blockDim.x + threadIdx.x;
  if (i < n) out[i] = x[i] + b_out[i % 6];
}

// One wave per (head h, group of 32 (b,n) chains). lane = chain.
// psi components live in 32 VGPRs; GP is 1024 fully-unrolled FMAs.
__global__ __launch_bounds__(32) void versor_scan(
    const float* __restrict__ x,      // [16,256,64,6]
    const float* __restrict__ W_in,   // [6,512]
    const float* __restrict__ b_in,   // [512]
    const float* __restrict__ W_out,  // [512,6]
    float* __restrict__ out)          // [16,256,64,6] (pre-init x+b_out)
{
  const int lane  = threadIdx.x;          // 0..31
  const int w     = blockIdx.x;           // 0..511
  const int h     = w & 15;               // head
  const int g     = w >> 4;               // chain group (32 chains each)
  const int lo16  = lane & 15;
  const bool loHalf = lane < 16;

  __shared__ float lds[32 * 33];          // u transpose buffer, padded stride

  // ---- B operand for in-projection WMMA: Win[h] as [K=d(0..5 pad 32), N=16 comps] f16.
  // VGPR v<3 holds rows K = 2v,2v+1 (lanes 0-15); lanes 16-31 and other VGPRs are zero.
  v16h Bwin[2];
  #pragma unroll
  for (int nt = 0; nt < 2; ++nt) {
    v16h b;
    #pragma unroll
    for (int e = 0; e < 16; ++e) b[e] = (_Float16)0.0f;
    #pragma unroll
    for (int v = 0; v < 3; ++v) {               // only K=0..5 nonzero, low half
      float f0 = loHalf ? W_in[(2 * v    ) * 512 + h * 32 + nt * 16 + lo16] : 0.0f;
      float f1 = loHalf ? W_in[(2 * v + 1) * 512 + h * 32 + nt * 16 + lo16] : 0.0f;
      b[2 * v]     = (_Float16)f0;
      b[2 * v + 1] = (_Float16)f1;
    }
    Bwin[nt] = b;
  }

  // b_in broadcast per comp (seed into the WMMA C accumulator)
  float binv[2];
  binv[0] = b_in[h * 32 + lo16];
  binv[1] = b_in[h * 32 + 16 + lo16];

  // ---- B operand for out-projection WMMA: Wout[h] as [K=32 comps, N=16 (d, pad)] f16.
  v16h Bwout;
  {
    const int d = lo16;
    #pragma unroll
    for (int v = 0; v < 8; ++v) {
      const int k = ((v < 4) ? 2 * v : 2 * v + 8) + (loHalf ? 0 : 8);
      float f0 = (d < 6) ? W_out[(h * 32 + k    ) * 6 + d] : 0.0f;
      float f1 = (d < 6) ? W_out[(h * 32 + k + 1) * 6 + d] : 0.0f;
      Bwout[2 * v]     = (_Float16)f0;
      Bwout[2 * v + 1] = (_Float16)f1;
    }
  }

  // ---- per-lane atomic output bases (step-invariant part)
  int abase[2][8];
  #pragma unroll
  for (int mt = 0; mt < 2; ++mt)
    #pragma unroll
    for (int r = 0; r < 8; ++r) {
      const int chain = mt * 16 + r + (loHalf ? 0 : 8);
      const int bn = g * 32 + chain;
      abase[mt][r] = (bn >> 6) * kSND + (bn & 63) * kD + lo16;  // d = lo16
    }

  // ---- per-lane x stream base
  const int bnL = g * 32 + lane;
  const float2* xp2 =
      (const float2*)(x + (bnL >> 6) * kSND + (bnL & 63) * kD);

  // ---- psi = e0
  float psi[32];
  #pragma unroll
  for (int k = 0; k < 32; ++k) psi[k] = 0.0f;
  psi[0] = 1.0f;

  for (int s = 0; s < kS; ++s) {
    // ---- load x_t for this lane's chain (24B, 8B aligned), prefetch next step
    float2 xa = xp2[s * (kND / 2) + 0];
    float2 xb = xp2[s * (kND / 2) + 1];
    float2 xc = xp2[s * (kND / 2) + 2];
    if (s + 1 < kS) __builtin_prefetch(&xp2[(s + 1) * (kND / 2)], 0, 3);
    float xv[6] = {xa.x, xa.y, xb.x, xb.y, xc.x, xc.y};

    // ---- in-projection: U[32 chains x 32 comps] = X[32x6] @ Win + b_in (4 WMMAs)
    v8f Cu[2][2];
    #pragma unroll
    for (int mt = 0; mt < 2; ++mt) {
      v16h Ax;
      #pragma unroll
      for (int e = 0; e < 16; ++e) Ax[e] = (_Float16)0.0f;
      #pragma unroll
      for (int v = 0; v < 3; ++v) {         // K=0..5 live in low lane-half
        float f0 = __shfl(xv[2 * v],     lo16 + mt * 16, 32);
        float f1 = __shfl(xv[2 * v + 1], lo16 + mt * 16, 32);
        Ax[2 * v]     = loHalf ? (_Float16)f0 : (_Float16)0.0f;
        Ax[2 * v + 1] = loHalf ? (_Float16)f1 : (_Float16)0.0f;
      }
      #pragma unroll
      for (int nt = 0; nt < 2; ++nt) {
        v8f c;
        #pragma unroll
        for (int r = 0; r < 8; ++r) c[r] = binv[nt];
        c = __builtin_amdgcn_wmma_f32_16x16x32_f16(
                false, Ax, false, Bwin[nt], (short)0, c, false, false);
        Cu[mt][nt] = c;
      }
    }

    // ---- transpose U through LDS (chain-major, stride 33: bank-conflict free)
    #pragma unroll
    for (int mt = 0; mt < 2; ++mt)
      #pragma unroll
      for (int nt = 0; nt < 2; ++nt)
        #pragma unroll
        for (int r = 0; r < 8; ++r) {
          const int chain = mt * 16 + r + (loHalf ? 0 : 8);
          const int comp  = nt * 16 + lo16;
          lds[chain * 33 + comp] = Cu[mt][nt][r];
        }
    // wave-private region; LDS ops are in-order within a wave

    float dr[32];
    #pragma unroll
    for (int k = 0; k < 32; ++k) dr[k] = lds[lane * 33 + k];
    dr[0] += 1.0f;

    // normalize delta_r (per-lane, in registers)
    float nrm = 1e-12f;
    #pragma unroll
    for (int k = 0; k < 32; ++k) nrm = fmaf(dr[k], dr[k], nrm);
    float rin = rsqrtf(nrm);
    #pragma unroll
    for (int k = 0; k < 32; ++k) dr[k] *= rin;

    // ---- geometric product: psi' = dr * psi  (1024 FMAs, signs folded)
    float pn[32];
    #pragma unroll
    for (int k = 0; k < 32; ++k) pn[k] = 0.0f;
    #pragma unroll
    for (int i = 0; i < 32; ++i)
      #pragma unroll
      for (int j = 0; j < 32; ++j)
        pn[i ^ j] = fmaf(kSign.s[i][j] * dr[i], psi[j], pn[i ^ j]);

    // normalize psi
    nrm = 1e-12f;
    #pragma unroll
    for (int k = 0; k < 32; ++k) nrm = fmaf(pn[k], pn[k], nrm);
    rin = rsqrtf(nrm);
    #pragma unroll
    for (int k = 0; k < 32; ++k) psi[k] = pn[k] * rin;

    // ---- out-projection: Y[32 chains x 6] = psi @ Wout (2 WMMAs, K=32 exact)
    #pragma unroll
    for (int mt = 0; mt < 2; ++mt) {
      v16h Ap;
      #pragma unroll
      for (int v = 0; v < 8; ++v) {
        const int kA = (v < 4) ? 2 * v : 2 * v + 8;   // low lane-half K
        const int kB = kA + 8;                        // high lane-half K
        float fa0 = __shfl(psi[kA],     lo16 + mt * 16, 32);
        float fa1 = __shfl(psi[kA + 1], lo16 + mt * 16, 32);
        float fb0 = __shfl(psi[kB],     lo16 + mt * 16, 32);
        float fb1 = __shfl(psi[kB + 1], lo16 + mt * 16, 32);
        Ap[2 * v]     = (_Float16)(loHalf ? fa0 : fb0);
        Ap[2 * v + 1] = (_Float16)(loHalf ? fa1 : fb1);
      }
      v8f cy;
      #pragma unroll
      for (int r = 0; r < 8; ++r) cy[r] = 0.0f;
      cy = __builtin_amdgcn_wmma_f32_16x16x32_f16(
              false, Ap, false, Bwout, (short)0, cy, false, false);

      // accumulate across the 16 head-waves via f32 atomics (out pre-seeded x+b_out)
      #pragma unroll
      for (int r = 0; r < 8; ++r)
        if (lo16 < 6)
          atomicAdd(&out[abase[mt][r] + s * kND], cy[r]);
    }
  }
}

extern "C" void kernel_launch(void* const* d_in, const int* in_sizes, int n_in,
                              void* d_out, int out_size, void* d_ws, size_t ws_size,
                              hipStream_t stream) {
  const float* x     = (const float*)d_in[0];
  const float* W_in  = (const float*)d_in[1];
  const float* b_in  = (const float*)d_in[2];
  const float* W_out = (const float*)d_in[3];
  const float* b_out = (const float*)d_in[4];
  float* out = (float*)d_out;

  const int n = out_size;  // 16*256*64*6 = 1572864
  versor_init_out<<<(n + 255) / 256, 256, 0, stream>>>(x, b_out, out, n);
  versor_scan<<<512, 32, 0, stream>>>(x, W_in, b_in, W_out, out);
}
